// LSTM_74586402062667
// MI455X (gfx1250) — matrix operand order repaired and checked by reading, not compiled, not run on hardware
//
#include <hip/hip_runtime.h>
#include <hip/hip_bf16.h>

// LSTM: SEQ=256, BATCH=64, HID=1024, LAYERS=2  (persistent-kernel version)
// One persistent kernel runs all 256 steps x 2 layers with device-scope
// barriers. Activation panels staged in LDS (async global->LDS when the
// builtin exists); hot loop = ds_load_b128 (A) + global b128 (B) + WMMA.

#define SEQ    256
#define BATCH  64
#define HID    1024
#define GH     (4 * HID)          // 4096 gate rows
#define BH     (BATCH * HID)      // 65536 elems per [B,H] slab
#define NWG    128                // 64 n-tiles x 2 batch-row groups
#define KCH    512                // K-chunk staged in LDS
#define APAD   8                  // panel row padding (keeps 16B align)

typedef __attribute__((ext_vector_type(16))) __bf16 v16bf;
typedef __attribute__((ext_vector_type(8)))  __bf16 v8bf;
typedef __attribute__((ext_vector_type(8)))  float  v8f;
typedef __attribute__((ext_vector_type(4)))  int    v4i;

#if __has_builtin(__builtin_amdgcn_global_load_async_to_lds_b128)
#define HAVE_ASYNC_LDS 1
#else
#define HAVE_ASYNC_LDS 0
#endif

__device__ __forceinline__ void wait_async_lds() {
#if HAVE_ASYNC_LDS
#if __has_builtin(__builtin_amdgcn_s_wait_asynccnt)
    __builtin_amdgcn_s_wait_asynccnt(0);
#else
    asm volatile("s_wait_asynccnt 0" ::: "memory");
#endif
#endif
}

// Stage 32 rows x KCH cols (bf16) of a [64,HID] panel into LDS.
__device__ __forceinline__ void stage_chunk(const __bf16* __restrict__ src,
                                            __bf16 (*pan)[KCH + APAD],
                                            int mgrp, int kh) {
    const int tid = threadIdx.x;
    #pragma unroll
    for (int i = 0; i < 16; ++i) {
        const int linear = i * 128 + tid;       // 0..2047 (16B chunks)
        const int row    = linear >> 6;         // 0..31
        const int chunk  = linear & 63;         // 0..63
        const __bf16* s = src + (size_t)(mgrp * 32 + row) * HID + kh + chunk * 8;
        __bf16*       d = &pan[row][chunk * 8];
#if HAVE_ASYNC_LDS
        __builtin_amdgcn_global_load_async_to_lds_b128((v4i*)s, (v4i*)d, 0, 0);
#else
        *(v8bf*)d = *(const v8bf*)s;
#endif
    }
    wait_async_lds();
}

// A fragment (16x32 bf16) from an LDS panel row.
// Lanes 0-15 (M=lane): elems 0-15 -> K=0..15
// Lanes 16-31 (M=l-16): elems 0-15 -> K=16..31
__device__ __forceinline__ v16bf lds_a_frag(const __bf16* __restrict__ rowp,
                                            int kk, int aoff) {
    v8bf lo = *(const v8bf*)(rowp + kk + aoff);
    v8bf hi = *(const v8bf*)(rowp + kk + 8 + aoff);
    v16bf r;
    #pragma unroll
    for (int i = 0; i < 8; ++i) { r[i] = lo[i]; r[8 + i] = hi[i]; }
    return r;
}

// Device-wide barrier: monotonically increasing arrival counter.
__device__ __forceinline__ void grid_barrier(unsigned* bar, unsigned target) {
    __syncthreads();
    if (threadIdx.x == 0) {
        __hip_atomic_fetch_add(bar, 1u, __ATOMIC_ACQ_REL, __HIP_MEMORY_SCOPE_AGENT);
        while (__hip_atomic_load(bar, __ATOMIC_ACQUIRE, __HIP_MEMORY_SCOPE_AGENT) < target)
            __builtin_amdgcn_s_sleep(1);
    }
    __syncthreads();
}

__device__ __forceinline__ float fast_sigmoid(float x) {
    return 1.0f / (1.0f + __expf(-x));
}
__device__ __forceinline__ float fast_tanh(float x) {
    const float xc = fminf(fmaxf(x, -20.0f), 20.0f);
    const float e  = __expf(-2.0f * xc);
    return (1.0f - e) / (1.0f + e);
}

// One GEMM half: gates += panel(LDS) @ W^T over K = kh..kh+KCH per chunk.
__device__ __forceinline__ void gemm_half(const __bf16* __restrict__ act,  // [64,HID]
                                          const __bf16* __restrict__ wrow, // lane's W row
                                          __bf16 (*pan)[KCH + APAD],
                                          int mgrp, int lm, int aoff, int boff,
                                          v8f& acc0, v8f& acc1) {
    for (int kh = 0; kh < HID; kh += KCH) {
        __syncthreads();                        // previous panel readers done
        stage_chunk(act, pan, mgrp, kh);
        __syncthreads();                        // panel visible to all waves
        const __bf16* r0 = pan[lm];
        const __bf16* r1 = pan[16 + lm];
        #pragma unroll 4
        for (int kk = 0; kk < KCH; kk += 32) {
            v16bf b  = *(const v16bf*)(wrow + kh + kk + boff);
            v16bf a0 = lds_a_frag(r0, kk, aoff);
            v16bf a1 = lds_a_frag(r1, kk, aoff);
            acc0 = __builtin_amdgcn_wmma_f32_16x16x32_bf16(false, a0, false, b,
                                                           (short)0, acc0, false, false);
            acc1 = __builtin_amdgcn_wmma_f32_16x16x32_bf16(false, a1, false, b,
                                                           (short)0, acc1, false, false);
        }
    }
}

// One LSTM layer step for this WG's (ntile, mgrp) tile.
__device__ __forceinline__ void lstm_layer_step(
    const __bf16* __restrict__ abf,     // [64,1024] layer input, bf16
    const __bf16* __restrict__ hbf,     // [64,1024] prev hidden, bf16
    const __bf16* __restrict__ Wxb,     // [4096,1024] bf16
    const __bf16* __restrict__ Whb,     // [4096,1024] bf16
    const float*  __restrict__ bsum,    // [4096] bx+bh
    float*        __restrict__ cbuf,    // [64,1024] cell state (in/out)
    __bf16*       __restrict__ hbf_out, // [64,1024] new hidden, bf16
    float*        __restrict__ ys,      // [64,1024] ys slice or nullptr
    float*        __restrict__ hT,      // final-h f32 dst or nullptr
    float*        __restrict__ cT,      // final-c f32 dst or nullptr
    int ntile, int mgrp, int wave, int lane,
    __bf16 (*pan)[KCH + APAD], float (*gsm)[32][16])
{
    const int lm   = lane & 15;
    const int aoff = (lane >> 4) * 16;
    const int boff = (lane >> 4) * 16;

    const int rowW = wave * HID + ntile * 16 + lm;
    const __bf16* wx_row = Wxb + (size_t)rowW * HID;
    const __bf16* wh_row = Whb + (size_t)rowW * HID;

    v8f acc0 = {}, acc1 = {};
    gemm_half(abf, wx_row, pan, mgrp, lm, aoff, boff, acc0, acc1);
    gemm_half(hbf, wh_row, pan, mgrp, lm, aoff, boff, acc0, acc1);

    // C/D layout: VGPR r -> M=r (lanes 0-15) / M=8+r (lanes 16-31); N=lane%16.
    __syncthreads();
    const int mbase = (lane >> 4) * 8;
    #pragma unroll
    for (int r = 0; r < 8; ++r) {
        gsm[wave][mbase + r][lm]      = acc0[r];
        gsm[wave][16 + mbase + r][lm] = acc1[r];
    }
    __syncthreads();

    // Elementwise LSTM cell: 32 rows x 16 cols = 512 elems over 128 threads.
    for (int e = threadIdx.x; e < 512; e += 128) {
        const int em  = e >> 4;
        const int en  = e & 15;
        const int col = ntile * 16 + en;
        const int row = mgrp * 32 + em;
        const float iv = gsm[0][em][en] + bsum[0 * HID + col];
        const float fv = gsm[1][em][en] + bsum[1 * HID + col];
        const float gv = gsm[2][em][en] + bsum[2 * HID + col];
        const float ov = gsm[3][em][en] + bsum[3 * HID + col];
        const float si = fast_sigmoid(iv);
        const float sf = fast_sigmoid(fv);
        const float so = fast_sigmoid(ov);
        const float tg = fast_tanh(gv);
        const size_t idx = (size_t)row * HID + col;
        const float cn = sf * cbuf[idx] + si * tg;
        const float hv = so * fast_tanh(cn);
        cbuf[idx]    = cn;
        hbf_out[idx] = (__bf16)hv;
        if (ys) ys[idx] = hv;
        if (hT) hT[idx] = hv;
        if (cT) cT[idx] = cn;
    }
}

// ---------------- persistent kernel: all steps, all layers -------------------
__global__ __launch_bounds__(128) void lstm_persistent(
    const __bf16* __restrict__ xbf,   // [256,64,1024] bf16
    const __bf16* __restrict__ Wxb,   // [2,4096,1024] bf16
    const __bf16* __restrict__ Whb,   // [2,4096,1024] bf16
    const float*  __restrict__ bsum,  // [2,4096]
    __bf16* h1a, __bf16* h1b,         // layer-1 hidden ping-pong (bf16)
    __bf16* h2a, __bf16* h2b,         // layer-2 hidden ping-pong (bf16)
    float* c1, float* c2,             // cell states (f32)
    float* ys_out,                    // [256,64,1024] f32 (d_out base)
    float* hT, float* cT,             // [2,64,1024] each (d_out tail)
    unsigned* bar)                    // device barrier counter (pre-zeroed)
{
    const int ntile = blockIdx.x & 63;
    const int mgrp  = blockIdx.x >> 6;
    const int wave  = threadIdx.x >> 5;
    const int lane  = threadIdx.x & 31;
    __shared__ __bf16 pan[32][KCH + APAD];   // 33 KB activation panel
    __shared__ float  gsm[4][32][16];        // 8 KB gate tiles

    const size_t L = (size_t)GH * HID;
    __bf16* h1[2] = { h1a, h1b };
    __bf16* h2[2] = { h2a, h2b };

    unsigned phase = 0;
    for (int t = 0; t < SEQ; ++t) {
        const int p = t & 1;
        const bool last = (t == SEQ - 1);
        // layer 1: input xbf[t] -> h1[1-p]
        lstm_layer_step(xbf + (size_t)t * BH, h1[p], Wxb, Whb, bsum,
                        c1, h1[1 - p], nullptr,
                        last ? hT : nullptr, last ? cT : nullptr,
                        ntile, mgrp, wave, lane, pan, gsm);
        grid_barrier(bar, (++phase) * NWG);
        // layer 2: input h1[1-p] -> h2[1-p], ys[t]
        lstm_layer_step(h1[1 - p], h2[p], Wxb + L, Whb + L, bsum + GH,
                        c2, h2[1 - p], ys_out + (size_t)t * BH,
                        last ? hT + BH : nullptr, last ? cT + BH : nullptr,
                        ntile, mgrp, wave, lane, pan, gsm);
        grid_barrier(bar, (++phase) * NWG);
    }
}

// ---------------- prep kernels ----------------------------------------------
__global__ void cvt_f32_to_bf16(const float* __restrict__ src,
                                __bf16* __restrict__ dst, int n) {
    int i = blockIdx.x * blockDim.x + threadIdx.x;
    if (i < n) dst[i] = (__bf16)src[i];
}

__global__ void bias_sum_kernel(const float* __restrict__ bx,
                                const float* __restrict__ bh,
                                float* __restrict__ bsum, int n) {
    int i = blockIdx.x * blockDim.x + threadIdx.x;
    if (i < n) bsum[i] = bx[i] + bh[i];
}

// ---------------- host-side orchestration ------------------------------------
extern "C" void kernel_launch(void* const* d_in, const int* in_sizes, int n_in,
                              void* d_out, int out_size, void* d_ws, size_t ws_size,
                              hipStream_t stream) {
    const float* x   = (const float*)d_in[0];   // [256,64,1024]
    const float* h0  = (const float*)d_in[1];   // [2,64,1024]
    const float* c0  = (const float*)d_in[2];   // [2,64,1024]
    const float* Wx  = (const float*)d_in[3];   // [2,4096,1024]
    const float* bx  = (const float*)d_in[4];   // [2,4096]
    const float* Wh  = (const float*)d_in[5];   // [2,4096,1024]
    const float* bh  = (const float*)d_in[6];   // [2,4096]
    float* out = (float*)d_out;                 // ys | hT | cT

    const size_t WCNT = (size_t)2 * GH * HID;   // elems per weight tensor
    const size_t XCNT = (size_t)SEQ * BH;       // elems in x
    char* ws = (char*)d_ws;
    size_t off = 0;
    __bf16* Wxb  = (__bf16*)(ws + off); off += WCNT * sizeof(__bf16);
    __bf16* Whb  = (__bf16*)(ws + off); off += WCNT * sizeof(__bf16);
    __bf16* xbf  = (__bf16*)(ws + off); off += XCNT * sizeof(__bf16);
    float*  bsum = (float*) (ws + off); off += (size_t)2 * GH * sizeof(float);
    __bf16* h1bf[2]; __bf16* h2bf[2];
    h1bf[0] = (__bf16*)(ws + off); off += BH * sizeof(__bf16);
    h1bf[1] = (__bf16*)(ws + off); off += BH * sizeof(__bf16);
    h2bf[0] = (__bf16*)(ws + off); off += BH * sizeof(__bf16);
    h2bf[1] = (__bf16*)(ws + off); off += BH * sizeof(__bf16);
    float* c1 = (float*)(ws + off); off += BH * sizeof(float);
    float* c2 = (float*)(ws + off); off += BH * sizeof(float);
    unsigned* bar = (unsigned*)(ws + off); off += 256;

    // Per-call prep (deterministic, graph-capture-safe).
    {
        const int thr = 256;
        cvt_f32_to_bf16<<<(int)((WCNT + thr - 1) / thr), thr, 0, stream>>>(Wx, Wxb, (int)WCNT);
        cvt_f32_to_bf16<<<(int)((WCNT + thr - 1) / thr), thr, 0, stream>>>(Wh, Whb, (int)WCNT);
        cvt_f32_to_bf16<<<(int)((XCNT + thr - 1) / thr), thr, 0, stream>>>(x, xbf, (int)XCNT);
        cvt_f32_to_bf16<<<(BH + thr - 1) / thr, thr, 0, stream>>>(h0,      h1bf[0], BH);
        cvt_f32_to_bf16<<<(BH + thr - 1) / thr, thr, 0, stream>>>(h0 + BH, h2bf[0], BH);
        bias_sum_kernel<<<(2 * GH + thr - 1) / thr, thr, 0, stream>>>(bx, bh, bsum, 2 * GH);
        (void)hipMemcpyAsync(c1, c0,      BH * sizeof(float), hipMemcpyDeviceToDevice, stream);
        (void)hipMemcpyAsync(c2, c0 + BH, BH * sizeof(float), hipMemcpyDeviceToDevice, stream);
        (void)hipMemsetAsync(bar, 0, sizeof(unsigned), stream);
    }

    const size_t YS = (size_t)SEQ * BH;
    lstm_persistent<<<dim3(NWG), dim3(128), 0, stream>>>(
        xbf, Wxb, Whb, bsum,
        h1bf[0], h1bf[1], h2bf[0], h2bf[1],
        c1, c2,
        out, out + YS, out + YS + 2 * BH,
        bar);
}
